// FPSNet_32581621908086
// MI455X (gfx1250) — matrix operand order repaired and checked by reading, not compiled, not run on hardware
//
#include <hip/hip_runtime.h>

// ---------------------------------------------------------------------------
// FPSNet fused kernel for gfx1250 (MI455X).
// Reference: h = stack(d, |x_m|, |x_n|) -> 3 gated bilinear layers (C=128)
//            -> softmax over n -> einsum against x -> 2x128 projection.
// Fused per-row-m with online softmax; GEMMs via v_wmma_f32_16x16x32_f16.
// ---------------------------------------------------------------------------

typedef __attribute__((ext_vector_type(16))) _Float16 v16h;
typedef __attribute__((ext_vector_type(8)))  _Float16 v8h;
typedef __attribute__((ext_vector_type(8)))  float    v8f;

#define N_PTS   768
#define C_DIM   128
#define TILE_N  64
#define WAVES   4
#define THREADS 128

union V16U { v16h v; v8h h[2]; };

__device__ __forceinline__ v16h make_frag(const _Float16* lo, const _Float16* hi) {
  V16U u;
  u.h[0] = *(const v8h*)lo;
  u.h[1] = *(const v8h*)hi;
  return u.v;
}

// --------------------------- prep: f32 -> f16 weights ----------------------
__global__ void fps_prep(const float* __restrict__ Wa2, const float* __restrict__ Wb2,
                         const float* __restrict__ Wa3, const float* __restrict__ Wb3,
                         _Float16* __restrict__ w16) {
  int i = blockIdx.x * blockDim.x + threadIdx.x;
  if (i < C_DIM * C_DIM) {
    w16[0 * 16384 + i] = (_Float16)Wa2[i];
    w16[1 * 16384 + i] = (_Float16)Wb2[i];
    w16[2 * 16384 + i] = (_Float16)Wa3[i];
    w16[3 * 16384 + i] = (_Float16)Wb3[i];
  }
}

// --------------------------- main fused kernel -----------------------------
__global__ __launch_bounds__(THREADS)
void fps_main(const float* __restrict__ x,
              const float* __restrict__ Wa1, const float* __restrict__ ba1,
              const float* __restrict__ Wb1, const float* __restrict__ bb1,
              const float* __restrict__ ba2, const float* __restrict__ bb2,
              const float* __restrict__ ba3, const float* __restrict__ bb3,
              const _Float16* __restrict__ w16,   // [4][128][128] f16: Wa2,Wb2,Wa3,Wb3
              float* __restrict__ partial)        // [768][3][128]
{
  __shared__ __attribute__((aligned(16))) _Float16 h1[TILE_N * C_DIM];  // 16 KB
  __shared__ __attribute__((aligned(16))) _Float16 h2[TILE_N * C_DIM];  // 16 KB
  __shared__ float sWa1[C_DIM][4];      // w0,w1,w2,bias
  __shared__ float sWb1[C_DIM][4];
  __shared__ float sB[4][C_DIM];        // ba2, bb2, ba3, bb3
  __shared__ float tx[TILE_N][3];
  __shared__ float tsq[TILE_N];
  __shared__ float tnrm[TILE_N];
  __shared__ float rec[C_DIM][WAVES][5]; // per-wave softmax records (m,s,t0,t1,t2)

  const int m    = blockIdx.x;
  const int tid  = threadIdx.x;
  const int wv   = tid >> 5;
  const int lane = tid & 31;
  const int lh   = lane >> 4;    // lane half (K-group select in WMMA layouts)
  const int l15  = lane & 15;

  // Query point m (uniform per block).
  const float xm0 = x[3 * m + 0], xm1 = x[3 * m + 1], xm2 = x[3 * m + 2];
  const float sqm  = xm0 * xm0 + xm1 * xm1 + xm2 * xm2;
  const float nrmm = sqrtf(sqm);

  // Stage layer-1 weights/biases and layer-2/3 biases (tid == channel).
  {
    const int c = tid;
    sWa1[c][0] = Wa1[3 * c + 0]; sWa1[c][1] = Wa1[3 * c + 1];
    sWa1[c][2] = Wa1[3 * c + 2]; sWa1[c][3] = ba1[c];
    sWb1[c][0] = Wb1[3 * c + 0]; sWb1[c][1] = Wb1[3 * c + 1];
    sWb1[c][2] = Wb1[3 * c + 2]; sWb1[c][3] = bb1[c];
    sB[0][c] = ba2[c]; sB[1][c] = bb2[c];
    sB[2][c] = ba3[c]; sB[3][c] = bb3[c];
  }

  const _Float16* Wa2h = w16 + 0 * 16384;
  const _Float16* Wb2h = w16 + 1 * 16384;
  const _Float16* Wa3h = w16 + 2 * 16384;
  const _Float16* Wb3h = w16 + 3 * 16384;

  // Running online-softmax state: this thread owns channel c == tid.
  float rM = -3.0e38f, rS = 0.f, rT0 = 0.f, rT1 = 0.f, rT2 = 0.f;

  const int arow = 16 * wv + l15;   // A-fragment row (wave-private strip)

  for (int nt = 0; nt < N_PTS / TILE_N; ++nt) {
    const int n0 = nt * TILE_N;

    // Stage the tile's points + norms.
    if (tid < TILE_N) {
      const float a = x[3 * (n0 + tid) + 0];
      const float b = x[3 * (n0 + tid) + 1];
      const float c = x[3 * (n0 + tid) + 2];
      tx[tid][0] = a; tx[tid][1] = b; tx[tid][2] = c;
      const float s = a * a + b * b + c * c;
      tsq[tid]  = s;
      tnrm[tid] = sqrtf(s);
    }
    __syncthreads();

    // Layer 1: each wave fills its own 16-row strip of h1 (f16 in LDS).
    for (int e = lane; e < 16 * C_DIM; e += 32) {
      const int rl = e >> 7;
      const int c  = e & 127;
      const int nl = 16 * wv + rl;
      float d2 = sqm + tsq[nl]
               - 2.f * (xm0 * tx[nl][0] + xm1 * tx[nl][1] + xm2 * tx[nl][2]);
      const float dd = d2 > 0.f ? sqrtf(d2) : 0.f;
      const float f2 = tnrm[nl];
      const float va = dd * sWa1[c][0] + nrmm * sWa1[c][1] + f2 * sWa1[c][2] + sWa1[c][3];
      const float vb = dd * sWb1[c][0] + nrmm * sWb1[c][1] + f2 * sWb1[c][2] + sWb1[c][3];
      h1[nl * C_DIM + c] = (_Float16)(va * vb);
    }
    // h1 strip is wave-private: no block barrier needed before consuming it.

    // Layer 2: h2 = (h1@Wa2^T + ba2) * (h1@Wb2^T + bb2), WMMA 16x16x32 f16.
    for (int ct = 0; ct < 8; ++ct) {
      const int cc = ct * 16 + l15;
      v8f accA = {}; v8f accB = {};
#pragma unroll
      for (int kk = 0; kk < 4; ++kk) {
        const int kb = kk * 32;
        // A (16x32 f16): lanes 0-15 hold K {kb..kb+7, kb+16..kb+23}, lanes 16-31 +8.
        v16h af = make_frag(&h1[arow * C_DIM + kb + 8 * lh],
                            &h1[arow * C_DIM + kb + 16 + 8 * lh]);
        // B (32x16 f16): lane column cc, lanes 0-15 hold K {kb..kb+7, kb+16..kb+23}, lanes 16-31 +8.
        v16h bfa = make_frag(&Wa2h[cc * C_DIM + kb + 8 * lh],
                             &Wa2h[cc * C_DIM + kb + 16 + 8 * lh]);
        v16h bfb = make_frag(&Wb2h[cc * C_DIM + kb + 8 * lh],
                             &Wb2h[cc * C_DIM + kb + 16 + 8 * lh]);
        accA = __builtin_amdgcn_wmma_f32_16x16x32_f16(false, af, false, bfa,
                                                      (short)0, accA, false, false);
        accB = __builtin_amdgcn_wmma_f32_16x16x32_f16(false, af, false, bfb,
                                                      (short)0, accB, false, false);
      }
      const float bA = sB[0][cc], bB = sB[1][cc];
#pragma unroll
      for (int j = 0; j < 8; ++j) {   // D layout: row = j + 8*lh, col = cc
        const float v = (accA[j] + bA) * (accB[j] + bB);
        h2[(16 * wv + j + 8 * lh) * C_DIM + cc] = (_Float16)v;
      }
    }

    // Layer 3 + online softmax over this tile's 64 n-rows.
    for (int ct = 0; ct < 8; ++ct) {
      const int cc = ct * 16 + l15;
      v8f accA = {}; v8f accB = {};
#pragma unroll
      for (int kk = 0; kk < 4; ++kk) {
        const int kb = kk * 32;
        v16h af = make_frag(&h2[arow * C_DIM + kb + 8 * lh],
                            &h2[arow * C_DIM + kb + 16 + 8 * lh]);
        v16h bfa = make_frag(&Wa3h[cc * C_DIM + kb + 8 * lh],
                             &Wa3h[cc * C_DIM + kb + 16 + 8 * lh]);
        v16h bfb = make_frag(&Wb3h[cc * C_DIM + kb + 8 * lh],
                             &Wb3h[cc * C_DIM + kb + 16 + 8 * lh]);
        accA = __builtin_amdgcn_wmma_f32_16x16x32_f16(false, af, false, bfa,
                                                      (short)0, accA, false, false);
        accB = __builtin_amdgcn_wmma_f32_16x16x32_f16(false, af, false, bfb,
                                                      (short)0, accB, false, false);
      }
      const float bA = sB[2][cc], bB = sB[3][cc];
      float h3v[8];
      float lm = -3.0e38f;
#pragma unroll
      for (int j = 0; j < 8; ++j) {
        h3v[j] = (accA[j] + bA) * (accB[j] + bB);
        lm = fmaxf(lm, h3v[j]);
      }
      float ls = 0.f, lt0 = 0.f, lt1 = 0.f, lt2 = 0.f;
#pragma unroll
      for (int j = 0; j < 8; ++j) {
        const int nl = 16 * wv + j + 8 * lh;
        const float e = __expf(h3v[j] - lm);
        ls  += e;
        lt0 += e * tx[nl][0];
        lt1 += e * tx[nl][1];
        lt2 += e * tx[nl][2];
      }
      // Merge with partner lane (lane ^ 16): same column, other 8 rows.
      const float om = __shfl_xor(lm,  16, 32);
      const float os = __shfl_xor(ls,  16, 32);
      const float o0 = __shfl_xor(lt0, 16, 32);
      const float o1 = __shfl_xor(lt1, 16, 32);
      const float o2 = __shfl_xor(lt2, 16, 32);
      const float nm = fmaxf(lm, om);
      const float fs = __expf(lm - nm), fo = __expf(om - nm);
      if (lh == 0) {
        rec[cc][wv][0] = nm;
        rec[cc][wv][1] = ls  * fs + os * fo;
        rec[cc][wv][2] = lt0 * fs + o0 * fo;
        rec[cc][wv][3] = lt1 * fs + o1 * fo;
        rec[cc][wv][4] = lt2 * fs + o2 * fo;
      }
    }
    __syncthreads();

    // Merge the 4 wave records into the running state (thread c = tid).
    {
      const int c = tid;
#pragma unroll
      for (int w2 = 0; w2 < WAVES; ++w2) {
        const float wm = rec[c][w2][0], ws = rec[c][w2][1];
        const float w0 = rec[c][w2][2], w1 = rec[c][w2][3], w2v = rec[c][w2][4];
        const float nm = fmaxf(rM, wm);
        const float fr = __expf(rM - nm), fw = __expf(wm - nm);
        rS  = rS  * fr + ws  * fw;
        rT0 = rT0 * fr + w0  * fw;
        rT1 = rT1 * fr + w1  * fw;
        rT2 = rT2 * fr + w2v * fw;
        rM  = nm;
      }
    }
    __syncthreads();   // rec/tx are rewritten next iteration
  }

  // agg contribution of row m: sum_n att[m,n,c] * x[n,i] = T_i / S
  const float inv = 1.0f / rS;
  partial[m * 384 + 0 * 128 + tid] = rT0 * inv;
  partial[m * 384 + 1 * 128 + tid] = rT1 * inv;
  partial[m * 384 + 2 * 128 + tid] = rT2 * inv;
}

// ---------------- reduce over m + final 2x128 projection -------------------
__global__ __launch_bounds__(384)
void fps_reduce(const float* __restrict__ partial, const float* __restrict__ x,
                const float* __restrict__ Wf, float* __restrict__ out) {
  __shared__ float agg[3][C_DIM];   // agg[i][c] = sum_m partial[m][i][c]
  const int t = threadIdx.x;        // 0..383
  const int i = t >> 7, c = t & 127;
  float s = 0.f;
  for (int m = 0; m < N_PTS; ++m) s += partial[m * 384 + i * 128 + c];
  agg[i][c] = s;
  __syncthreads();
  if (t < 3) out[t] = x[t];                       // x[:, :1, :]
  if (t < 6) {                                    // out[o,d] = sum_c Wf[o,c]*agg[d][c]
    const int o = t / 3, d = t % 3;
    float acc = 0.f;
    for (int cc = 0; cc < C_DIM; ++cc) acc += Wf[o * C_DIM + cc] * agg[d][cc];
    out[3 + o * 3 + d] = acc;
  }
}

// ---------------------------------------------------------------------------
extern "C" void kernel_launch(void* const* d_in, const int* in_sizes, int n_in,
                              void* d_out, int out_size, void* d_ws, size_t ws_size,
                              hipStream_t stream) {
  (void)in_sizes; (void)n_in; (void)out_size; (void)ws_size;
  const float* x   = (const float*)d_in[0];
  const float* Wa1 = (const float*)d_in[1];
  const float* ba1 = (const float*)d_in[2];
  const float* Wb1 = (const float*)d_in[3];
  const float* bb1 = (const float*)d_in[4];
  const float* Wa2 = (const float*)d_in[5];
  const float* ba2 = (const float*)d_in[6];
  const float* Wb2 = (const float*)d_in[7];
  const float* bb2 = (const float*)d_in[8];
  const float* Wa3 = (const float*)d_in[9];
  const float* ba3 = (const float*)d_in[10];
  const float* Wb3 = (const float*)d_in[11];
  const float* bb3 = (const float*)d_in[12];
  const float* Wf  = (const float*)d_in[13];
  float* out = (float*)d_out;

  // Workspace layout: [4*128*128 f16 weights][768*3*128 f32 partials]
  _Float16* w16    = (_Float16*)d_ws;
  float*    partial = (float*)((char*)d_ws + 4 * C_DIM * C_DIM * sizeof(_Float16));

  fps_prep<<<(C_DIM * C_DIM + 255) / 256, 256, 0, stream>>>(Wa2, Wb2, Wa3, Wb3, w16);
  fps_main<<<N_PTS, THREADS, 0, stream>>>(x, Wa1, ba1, Wb1, bb1,
                                          ba2, bb2, ba3, bb3, w16, partial);
  fps_reduce<<<1, 384, 0, stream>>>(partial, x, Wf, out);
}
